// localUp_60851096650194
// MI455X (gfx1250) — matrix operand order repaired and checked by reading, not compiled, not run on hardware
//
#include <hip/hip_runtime.h>
#include <hip/hip_bf16.h>

typedef __attribute__((ext_vector_type(16))) _Float16 v16h;
typedef __attribute__((ext_vector_type(8)))  _Float16 v8h;
typedef __attribute__((ext_vector_type(8)))  float    v8f;
typedef __attribute__((ext_vector_type(4)))  float    f4;

namespace {
constexpr int kHW  = 128 * 128;   // 16384 pixels per image plane
constexpr int kC1  = 512;
constexpr int kC2  = 1024;
constexpr int kCO  = 256;
constexpr int kKD  = 64;
constexpr int kSHW = 64 * 64;     // 4096, source plane for c2 / out_feat
constexpr float kUPS = 63.0f / 127.0f;   // linspace(0,63,128) step
}

// ---------------------------------------------------------------------------
// Prep: convert w1/w2 to f16, fold BN into scale/bias.
// ---------------------------------------------------------------------------
__global__ __launch_bounds__(256) void prep_kernel(
    const float* __restrict__ w1, const float* __restrict__ w2,
    const float* __restrict__ g1, const float* __restrict__ be1,
    const float* __restrict__ m1, const float* __restrict__ v1,
    const float* __restrict__ g2, const float* __restrict__ be2,
    const float* __restrict__ m2, const float* __restrict__ v2,
    _Float16* __restrict__ w1h, _Float16* __restrict__ w2h,
    float* __restrict__ s1, float* __restrict__ bb1,
    float* __restrict__ s2, float* __restrict__ bb2)
{
  const int gid = blockIdx.x * 256 + threadIdx.x;
  if (gid < kKD * kC1) w1h[gid] = (_Float16)w1[gid];
  if (gid < kKD * kC2) w2h[gid] = (_Float16)w2[gid];
  if (gid < kKD) {
    float a = g1[gid] * rsqrtf(v1[gid] + 1e-5f);
    s1[gid] = a;  bb1[gid] = be1[gid] - m1[gid] * a;
    float b = g2[gid] * rsqrtf(v2[gid] + 1e-5f);
    s2[gid] = b;  bb2[gid] = be2[gid] - m2[gid] * b;
  }
}

// ---------------------------------------------------------------------------
// GEMM1: q[pixel][kd] = BN( w1 (64x512) x c1 (512 x pixels) ), f16 out.
// One wave = one 16-pixel tile, 4 M-tiles (KD=64), K stepped by 32.
// ---------------------------------------------------------------------------
__global__ __launch_bounds__(256) void gemm_q_kernel(
    const float* __restrict__ c1, const _Float16* __restrict__ w1h,
    const float* __restrict__ s1, const float* __restrict__ bb1,
    _Float16* __restrict__ qws)
{
  const int lane  = threadIdx.x & 31;
  const int wave  = threadIdx.x >> 5;
  const int tile  = blockIdx.x * 8 + wave;       // 4096 tiles
  const int pix0  = tile << 4;
  const int n_img = pix0 >> 14;                  // /kHW
  const int prem  = pix0 & (kHW - 1);
  const float* bbase = c1 + (size_t)n_img * kC1 * kHW + prem;
  const int hi8  = (lane & 16) ? 8 : 0;
  const int mrow = lane & 15;

  v8f acc[4] = {v8f{}, v8f{}, v8f{}, v8f{}};

  for (int kb = 0; kb < kC1; kb += 32) {
    // B fragment: lane -> pixel column (lane&15), halves 0..7 = K kb+hi8..+7,
    // halves 8..15 = K kb+16+hi8..+7
    const float* src = bbase + mrow;
    v16h bm;
#pragma unroll
    for (int i = 0; i < 16; ++i) {
      const int kk = (i < 8) ? (hi8 + i) : (8 + hi8 + i);
      bm[i] = (_Float16)src[(size_t)(kb + kk) * kHW];
    }
#pragma unroll
    for (int mt = 0; mt < 4; ++mt) {
      // A fragment (16-bit A 16x32 layout): halves 0..7 = K kb+hi8..+7,
      // halves 8..15 = K kb+16+hi8..+7
      const _Float16* wr = w1h + (size_t)(mt * 16 + mrow) * kC1 + kb;
      v8h lo = *(const v8h*)(wr + hi8);
      v8h hv = *(const v8h*)(wr + 16 + hi8);
      v16h am;
#pragma unroll
      for (int i = 0; i < 8; ++i) { am[i] = lo[i]; am[i + 8] = hv[i]; }
      acc[mt] = __builtin_amdgcn_wmma_f32_16x16x32_f16(
          false, am, false, bm, (short)0, acc[mt], false, false);
    }
  }

  // D layout: lane = pixel column (lane&15), VGPR r = M row (+8 for hi lanes)
  const int ncol = lane & 15;
  _Float16* qrow = qws + (size_t)(pix0 + ncol) * kKD;
#pragma unroll
  for (int mt = 0; mt < 4; ++mt) {
    const int m0 = mt * 16 + hi8;
    v8h o;
#pragma unroll
    for (int r = 0; r < 8; ++r)
      o[r] = (_Float16)(acc[mt][r] * s1[m0 + r] + bb1[m0 + r]);
    *(v8h*)(qrow + m0) = o;
  }
}

// ---------------------------------------------------------------------------
// GEMM2: k[pixel][kd] = BN( w2 (64x1024) x bilinear_up(c2) ), upsample fused
// into the B-fragment build (each 16-pixel tile lives in one output row).
// ---------------------------------------------------------------------------
__global__ __launch_bounds__(256) void gemm_k_kernel(
    const float* __restrict__ c2, const _Float16* __restrict__ w2h,
    const float* __restrict__ s2, const float* __restrict__ bb2,
    _Float16* __restrict__ kws)
{
  const int lane  = threadIdx.x & 31;
  const int wave  = threadIdx.x >> 5;
  const int tile  = blockIdx.x * 8 + wave;       // 4096 tiles
  const int pix0  = tile << 4;
  const int n_img = pix0 >> 14;
  const int prem  = pix0 & (kHW - 1);
  const int yrow  = prem >> 7;
  const int xbase = prem & 127;
  const int hi8  = (lane & 16) ? 8 : 0;
  const int mrow = lane & 15;

  // Bilinear coefficients (constant over the K loop)
  const float yy = yrow * kUPS;
  const int   y0 = (int)yy;
  const float wy = yy - (float)y0;
  const int   y1 = (y0 + 1 < 63) ? (y0 + 1) : 63;
  int   px0[16], px1[16];
  float pwx[16];
#pragma unroll
  for (int i = 0; i < 16; ++i) {
    float xx = (float)(xbase + i) * kUPS;
    int   x0 = (int)xx;
    px0[i] = x0;
    px1[i] = (x0 + 1 < 63) ? (x0 + 1) : 63;
    pwx[i] = xx - (float)x0;
  }

  v8f acc[4] = {v8f{}, v8f{}, v8f{}, v8f{}};

  for (int kb = 0; kb < kC2; kb += 32) {
    v16h bm;
#pragma unroll
    for (int i = 0; i < 16; ++i) {
      const int kk = (i < 8) ? (hi8 + i) : (8 + hi8 + i);
      const float* cb = c2 + (size_t)(n_img * kC2 + kb + kk) * kSHW;
      const float* r0 = cb + y0 * 64;
      const float* r1 = cb + y1 * 64;
      float wx = pwx[mrow];
      float v0 = r0[px0[mrow]] * (1.0f - wx) + r0[px1[mrow]] * wx;
      float v1 = r1[px0[mrow]] * (1.0f - wx) + r1[px1[mrow]] * wx;
      bm[i] = (_Float16)(v0 * (1.0f - wy) + v1 * wy);
    }
#pragma unroll
    for (int mt = 0; mt < 4; ++mt) {
      const _Float16* wr = w2h + (size_t)(mt * 16 + mrow) * kC2 + kb;
      v8h lo = *(const v8h*)(wr + hi8);
      v8h hv = *(const v8h*)(wr + 16 + hi8);
      v16h am;
#pragma unroll
      for (int i = 0; i < 8; ++i) { am[i] = lo[i]; am[i + 8] = hv[i]; }
      acc[mt] = __builtin_amdgcn_wmma_f32_16x16x32_f16(
          false, am, false, bm, (short)0, acc[mt], false, false);
    }
  }

  const int ncol = lane & 15;
  _Float16* krow = kws + (size_t)(pix0 + ncol) * kKD;
#pragma unroll
  for (int mt = 0; mt < 4; ++mt) {
    const int m0 = mt * 16 + hi8;
    v8h o;
#pragma unroll
    for (int r = 0; r < 8; ++r)
      o[r] = (_Float16)(acc[mt][r] * s2[m0 + r] + bb2[m0 + r]);
    *(v8h*)(krow + m0) = o;
  }
}

// ---------------------------------------------------------------------------
// Upsample out_feat (4,256,64,64) -> ou (4,256,128,128) f16.
// ---------------------------------------------------------------------------
__global__ __launch_bounds__(256) void upsample_ou_kernel(
    const float* __restrict__ of, _Float16* __restrict__ ou)
{
  const int gid = blockIdx.x * 256 + threadIdx.x;   // 16,777,216 elements
  const int x = gid & 127;
  const int y = (gid >> 7) & 127;
  const int plane = gid >> 14;                      // n*256 + c
  const float yy = y * kUPS;
  const int   y0 = (int)yy;
  const float wy = yy - (float)y0;
  const int   y1 = (y0 + 1 < 63) ? (y0 + 1) : 63;
  const float xx = x * kUPS;
  const int   x0 = (int)xx;
  const float wx = xx - (float)x0;
  const int   x1 = (x0 + 1 < 63) ? (x0 + 1) : 63;
  const float* b = of + (size_t)plane * kSHW;
  float v = (b[y0 * 64 + x0] * (1.0f - wx) + b[y0 * 64 + x1] * wx) * (1.0f - wy) +
            (b[y1 * 64 + x0] * (1.0f - wx) + b[y1 * 64 + x1] * wx) * wy;
  ou[gid] = (_Float16)v;
}

// ---------------------------------------------------------------------------
// Attention: energies (9 dilated neighbors, K=64 dot), softmax, weighted
// 256-channel gather-sum.  Block = 64 consecutive pixels, 256 threads.
// ---------------------------------------------------------------------------
__global__ __launch_bounds__(256) void attn_kernel(
    const _Float16* __restrict__ qws, const _Float16* __restrict__ kws,
    const _Float16* __restrict__ ou, float* __restrict__ out)
{
  __shared__ float e_s[64][12];
  __shared__ float att_s[64][12];

  const int t   = threadIdx.x;
  const int px  = t & 63;
  const int grp = t >> 6;                 // 0..3
  const int p   = blockIdx.x * 64 + px;
  const int n_img = p >> 14;
  const int rem   = p & (kHW - 1);
  const int y = rem >> 7, x = rem & 127;

  // Phase 1: energies, 4-way split over j across the thread groups.
  v8h qv[8];
#pragma unroll
  for (int i = 0; i < 8; ++i)
    qv[i] = *(const v8h*)(qws + (size_t)p * kKD + i * 8);

  for (int j = grp; j < 9; j += 4) {
    const int dy = (j / 3) * 2 - 2;
    const int dx = (j % 3) * 2 - 2;
    const int ny = y + dy, nx = x + dx;
    float e = 0.0f;
    if (ny >= 0 && ny < 128 && nx >= 0 && nx < 128) {
      const _Float16* kr = kws + (size_t)(p + dy * 128 + dx) * kKD;
#pragma unroll
      for (int i = 0; i < 8; ++i) {
        v8h kv = *(const v8h*)(kr + i * 8);
#pragma unroll
        for (int u = 0; u < 8; ++u)
          e += (float)qv[i][u] * (float)kv[u];
      }
    }
    e_s[px][j] = e;
  }
  __syncthreads();

  if (t < 64) {
    float m = -1e30f;
#pragma unroll
    for (int j = 0; j < 9; ++j) m = fmaxf(m, e_s[t][j]);
    float a[9], s = 0.0f;
#pragma unroll
    for (int j = 0; j < 9; ++j) { a[j] = __expf(e_s[t][j] - m); s += a[j]; }
    const float inv = 1.0f / s;
#pragma unroll
    for (int j = 0; j < 9; ++j) att_s[t][j] = a[j] * inv;
  }
  __syncthreads();

  // Phase 2: out[c] = sum_j att[j] * ou[c][p + off_j]; one channel per thread (4 per block)
  // per sweep, pixels contiguous across lanes for coalescing.
  int   off[9];
  float aw[9];
#pragma unroll
  for (int j = 0; j < 9; ++j) {
    const int dy = (j / 3) * 2 - 2;
    const int dx = (j % 3) * 2 - 2;
    const int ny = y + dy, nx = x + dx;
    const bool ok = (ny >= 0 && ny < 128 && nx >= 0 && nx < 128);
    off[j] = ok ? (dy * 128 + dx) : 0;
    aw[j]  = ok ? att_s[px][j] : 0.0f;
  }

  for (int c = grp; c < kCO; c += 4) {
    const size_t pbase = (size_t)(n_img * kCO + c) * kHW + rem;
    const _Float16* op = ou + pbase;
    float acc = 0.0f;
#pragma unroll
    for (int j = 0; j < 9; ++j)
      acc += aw[j] * (float)op[off[j]];
    out[pbase] = acc;
  }
}

// ---------------------------------------------------------------------------
// Launch
// ---------------------------------------------------------------------------
extern "C" void kernel_launch(void* const* d_in, const int* in_sizes, int n_in,
                              void* d_out, int out_size, void* d_ws, size_t ws_size,
                              hipStream_t stream) {
  const float* c1  = (const float*)d_in[0];
  const float* c2  = (const float*)d_in[1];
  const float* of  = (const float*)d_in[2];
  const float* w1  = (const float*)d_in[3];
  const float* g1  = (const float*)d_in[4];
  const float* be1 = (const float*)d_in[5];
  const float* m1  = (const float*)d_in[6];
  const float* v1  = (const float*)d_in[7];
  const float* w2  = (const float*)d_in[8];
  const float* g2  = (const float*)d_in[9];
  const float* be2 = (const float*)d_in[10];
  const float* m2  = (const float*)d_in[11];
  const float* v2  = (const float*)d_in[12];

  char* ws = (char*)d_ws;
  _Float16* qws = (_Float16*)(ws);                    //  8,388,608 B
  _Float16* kws = (_Float16*)(ws + 8388608);          //  8,388,608 B
  _Float16* ouw = (_Float16*)(ws + 16777216);         // 33,554,432 B
  _Float16* w1h = (_Float16*)(ws + 50331648);         //     65,536 B
  _Float16* w2h = (_Float16*)(ws + 50397184);         //    131,072 B
  float*    s1  = (float*)   (ws + 50528256);
  float*    bb1 = s1 + 64;
  float*    s2  = s1 + 128;
  float*    bb2 = s1 + 192;

  prep_kernel<<<256, 256, 0, stream>>>(w1, w2, g1, be1, m1, v1,
                                       g2, be2, m2, v2,
                                       w1h, w2h, s1, bb1, s2, bb2);
  gemm_q_kernel<<<512, 256, 0, stream>>>(c1, w1h, s1, bb1, qws);
  gemm_k_kernel<<<512, 256, 0, stream>>>(c2, w2h, s2, bb2, kws);
  upsample_ou_kernel<<<65536, 256, 0, stream>>>(of, ouw);
  attn_kernel<<<1024, 256, 0, stream>>>(qws, kws, ouw, (float*)d_out);
}
